// GumbelVectorQuantizer_45406394253331
// MI455X (gfx1250) — matrix operand order repaired and checked by reading, not compiled, not run on hardware
//
#include <hip/hip_runtime.h>
#include <hip/hip_bf16.h>

// ---------------------------------------------------------------------------
// Gumbel Vector Quantizer forward for MI455X (gfx1250, wave32, WMMA).
// B=16 N=2048 E=1024 G=2 V=320 D=512 TAU=2.0
//   h = x@W + b                (bf16 WMMA 16x16x32 GEMM, 32768x1024x640,
//                               A converted f32->bf16 inline: x read once)
//   soft = softmax((h+g)/TAU)  (fused in GEMM epilogue via LDS + wave shfl)
//   y = hard one-hot (ST)      -> code_vector = y_sel * codebook[g, argmax]
//   entropy from accumulated selection weights
// HBM traffic ~ x 128MB + u 80MB + out 128MB ~= 336MB -> ~14.5us floor.
// B fragments are loaded 5-deep before the WMMA chain so L2 latency on the
// packed-W tiles overlaps the matrix pipe instead of serializing it.
// ---------------------------------------------------------------------------

typedef __bf16       v16bf __attribute__((ext_vector_type(16)));
typedef float        v8f   __attribute__((ext_vector_type(8)));
typedef unsigned int v8u   __attribute__((ext_vector_type(8)));

#define B_  16
#define N_  2048
#define E_  1024
#define G_  2
#define V_  320
#define D_  512
#define GV  640            // G*V columns of the projection
#define M_  (B_ * N_)      // 32768 rows
#define KCHUNKS (E_ / 32)  // 32 WMMA K-steps
#define NTILES  (GV / 16)  // 40 column tiles

// byte strides of the packed-W fragment array wp[tile][kc][lane][16 bf16]
#define WP_KC_STRIDE   1024    // 32 lanes * 16 el * 2 B
#define WP_TILE_STRIDE 32768   // KCHUNKS * WP_KC_STRIDE

// workspace layout (bytes)
#define WS_COUNTS 0            // 10240 f32   (B*G*V)
#define WS_WP     65536        // 655360 bf16 (packed W fragments)

__device__ __forceinline__ unsigned short f2bf(float f) {
  unsigned u = __builtin_bit_cast(unsigned, f);
  u += 0x7FFFu + ((u >> 16) & 1u);            // round-to-nearest-even
  return (unsigned short)(u >> 16);
}

// ---------------------------------------------------------------------------
__global__ void zero_counts_kernel(float* __restrict__ counts) {
  counts[blockIdx.x * 256 + threadIdx.x] = 0.0f;   // exactly 10240 threads
}

// W (f32, [E, GV] row-major) -> bf16 in exact per-lane B-fragment layout:
// wp[tile][kc][lane][e], lane holds column n = tile*16 + (lane&15),
// K = kc*32 + (lane>=16 ? 8 : 0) + (e&7) + (e>=8 ? 16 : 0)   (ISA table order).
__global__ void pack_w_kernel(const float* __restrict__ W,
                              unsigned short* __restrict__ wp) {
  int o    = blockIdx.x * blockDim.x + threadIdx.x;   // < 655360
  int e    = o & 15;
  int lane = (o >> 4) & 31;
  int kc   = (o >> 9) & 31;
  int tile = o >> 14;
  int n = tile * 16 + (lane & 15);
  int k = kc * 32 + ((lane >> 4) * 8) + (e & 7) + ((e >> 3) * 16);
  wp[o] = f2bf(W[(size_t)k * GV + n]);
}

// ---------------------------------------------------------------------------
// Fused: WMMA GEMM (16 rows x 640 cols per block) + gumbel + softmax/argmax
//        + codebook gather + selection-weight accumulation.
__global__ void __launch_bounds__(256)
gemm_vq_kernel(const float* __restrict__ x,
               const unsigned short* __restrict__ wp,
               const float* __restrict__ bias,
               const float* __restrict__ u,
               const float* __restrict__ codebook,
               float* __restrict__ out,
               float* __restrict__ counts) {
  __shared__ float h_lds[16 * GV];                 // 40 KB

  const int tid    = threadIdx.x;
  const int wave   = tid >> 5;
  const int lane   = tid & 31;
  const int laneLo = lane & 15;
  const int laneHi = lane >> 4;                    // 0 or 1
  const int m0     = blockIdx.x * 16;
  const int c0     = wave * 80;                    // 5 tiles of 16 cols

  v8f acc[5];
  const v8f vzero = {0.f, 0.f, 0.f, 0.f, 0.f, 0.f, 0.f, 0.f};
  #pragma unroll
  for (int t = 0; t < 5; ++t) acc[t] = vzero;

  // A base: row m0+laneLo, K sub-chunk per half-wave (ISA 16-bit A layout:
  // lanes 0-15 hold K [0..7,16..23], lanes 16-31 hold K [8..15,24..31]).
  const char* ap = (const char*)x + ((size_t)(m0 + laneLo) * E_) * 4
                   + (size_t)laneHi * 32;
  // B base: this wave's first tile, this lane's 16 contiguous bf16.
  const char* bp = (const char*)wp + (size_t)wave * 5 * WP_TILE_STRIDE
                   + (size_t)lane * 32;

  #pragma unroll 2
  for (int kc = 0; kc < KCHUNKS; ++kc) {
    // ---- issue ALL loads for this K-step up front (A: 4x b128, B: 10x b128)
    const char* akc = ap + kc * 128;
    float4 a0 = *reinterpret_cast<const float4*>(akc);        // K +0..3
    float4 a1 = *reinterpret_cast<const float4*>(akc + 16);   // K +4..7
    float4 a2 = *reinterpret_cast<const float4*>(akc + 64);   // K +16..19
    float4 a3 = *reinterpret_cast<const float4*>(akc + 80);   // K +20..23

    const char* bkc = bp + kc * WP_KC_STRIDE;
    v16bf barr[5];
    #pragma unroll
    for (int t = 0; t < 5; ++t) {
      const uint4 blo = *reinterpret_cast<const uint4*>(bkc + t * WP_TILE_STRIDE);
      const uint4 bhi = *reinterpret_cast<const uint4*>(bkc + t * WP_TILE_STRIDE + 16);
      v8u braw = {blo.x, blo.y, blo.z, blo.w, bhi.x, bhi.y, bhi.z, bhi.w};
      barr[t] = __builtin_bit_cast(v16bf, braw);
    }

    // ---- convert A to bf16 fragment (v_cvt_pk_bf16_f32 x8)
    v16bf af;
    af[0]  = (__bf16)a0.x; af[1]  = (__bf16)a0.y;
    af[2]  = (__bf16)a0.z; af[3]  = (__bf16)a0.w;
    af[4]  = (__bf16)a1.x; af[5]  = (__bf16)a1.y;
    af[6]  = (__bf16)a1.z; af[7]  = (__bf16)a1.w;
    af[8]  = (__bf16)a2.x; af[9]  = (__bf16)a2.y;
    af[10] = (__bf16)a2.z; af[11] = (__bf16)a2.w;
    af[12] = (__bf16)a3.x; af[13] = (__bf16)a3.y;
    af[14] = (__bf16)a3.z; af[15] = (__bf16)a3.w;

    // ---- WMMA chain; tile t only needs barr[t], later B loads still in flight
    #pragma unroll
    for (int t = 0; t < 5; ++t)
      acc[t] = __builtin_amdgcn_wmma_f32_16x16x32_bf16(
          false, af, false, barr[t], (short)0, acc[t], false, false);
  }

  // Epilogue: add bias, scatter C (lane holds col laneLo, rows r + laneHi*8)
  #pragma unroll
  for (int t = 0; t < 5; ++t) {
    int col = c0 + t * 16 + laneLo;
    float bc = bias[col];
    #pragma unroll
    for (int r = 0; r < 8; ++r)
      h_lds[(r + laneHi * 8) * GV + col] = acc[t][r] + bc;
  }
  __syncthreads();

  // Phase 2: 32 (row,group) tasks; each wave handles 4; 10 V-values per lane.
  for (int i = 0; i < 4; ++i) {
    int tau = wave * 4 + i;
    int r   = tau >> 1;
    int g2  = tau & 1;
    size_t m = (size_t)(m0 + r);
    int bb  = (int)(m >> 11);                    // m / N_

    const float* up = u + m * GV + (size_t)g2 * V_;
    const float* hp = h_lds + r * GV + g2 * V_;

    float tv[10];
    float mx = -__builtin_inff();
    int   mi = 0;
    #pragma unroll
    for (int j = 0; j < 10; ++j) {
      int v = lane + j * 32;
      float uu = up[v];
      float gg = -logf(-logf(fmaf(uu, 1.0f - 2e-7f, 1e-7f)));  // gumbel
      float t  = (hp[v] + gg) * 0.5f;                          // /TAU
      tv[j] = t;
      if (t > mx) { mx = t; mi = v; }
    }
    // wave32 butterfly argmax (first-occurrence tie rule: smaller index wins)
    #pragma unroll
    for (int off = 16; off > 0; off >>= 1) {
      float om = __shfl_xor(mx, off, 32);
      int   oi = __shfl_xor(mi, off, 32);
      if (om > mx || (om == mx && oi < mi)) { mx = om; mi = oi; }
    }
    float se = 0.0f;
    #pragma unroll
    for (int j = 0; j < 10; ++j) se += expf(tv[j] - mx);
    #pragma unroll
    for (int off = 16; off > 0; off >>= 1) se += __shfl_xor(se, off, 32);

    float s    = 1.0f / se;            // softmax prob at argmax
    float ysel = (1.0f - s) + s;       // straight-through forward value

    // code_vector[m, g2*D .. +D) = ysel * codebook[g2, mi, :]
    const float4* cbp = reinterpret_cast<const float4*>(
        codebook + ((size_t)g2 * V_ + mi) * D_);
    float4* op = reinterpret_cast<float4*>(out + m * (G_ * D_) + (size_t)g2 * D_);
    #pragma unroll
    for (int j = 0; j < 4; ++j) {
      float4 c4 = cbp[lane + j * 32];
      float4 o4 = make_float4(c4.x * ysel, c4.y * ysel, c4.z * ysel, c4.w * ysel);
      op[lane + j * 32] = o4;
    }
    if (lane == 0)
      atomicAdd(&counts[(bb * G_ + g2) * V_ + mi], ysel);
  }
}

// ---------------------------------------------------------------------------
// entropy = -sum_{b,g,v} p*log(p+1e-8) / (G*V),  p = softmax_V(counts[b,g,:])
__global__ void entropy_kernel(const float* __restrict__ counts,
                               float* __restrict__ out) {
  __shared__ float red[512];
  int tid = threadIdx.x;
  float accum = 0.0f;

  for (int bg = 0; bg < B_ * G_; ++bg) {
    float c = (tid < V_) ? counts[bg * V_ + tid] : -__builtin_inff();
    red[tid] = c;
    __syncthreads();
    for (int s = 256; s > 0; s >>= 1) {
      if (tid < s) red[tid] = fmaxf(red[tid], red[tid + s]);
      __syncthreads();
    }
    float mx = red[0];
    __syncthreads();
    float e = (tid < V_) ? expf(c - mx) : 0.0f;
    red[tid] = e;
    __syncthreads();
    for (int s = 256; s > 0; s >>= 1) {
      if (tid < s) red[tid] += red[tid + s];
      __syncthreads();
    }
    float tot = red[0];
    __syncthreads();
    if (tid < V_) {
      float p = e / tot;
      accum += p * logf(p + 1e-8f);
    }
  }

  red[tid] = accum;
  __syncthreads();
  for (int s = 256; s > 0; s >>= 1) {
    if (tid < s) red[tid] += red[tid + s];
    __syncthreads();
  }
  if (tid == 0)
    out[(size_t)M_ * (G_ * D_)] = -red[0] / (float)(G_ * V_);
}

// ---------------------------------------------------------------------------
extern "C" void kernel_launch(void* const* d_in, const int* in_sizes, int n_in,
                              void* d_out, int out_size, void* d_ws, size_t ws_size,
                              hipStream_t stream) {
  const float* x  = (const float*)d_in[0];   // [B,N,E]
  const float* u  = (const float*)d_in[1];   // [B,N,G,V]
  const float* W  = (const float*)d_in[2];   // [E, G*V]
  const float* b  = (const float*)d_in[3];   // [G*V]
  const float* cb = (const float*)d_in[4];   // [G,V,D]
  float* out = (float*)d_out;                // [B,N,G*D] ++ [1]

  char* ws = (char*)d_ws;
  float*          counts = (float*)(ws + WS_COUNTS);
  unsigned short* wp     = (unsigned short*)(ws + WS_WP);

  zero_counts_kernel<<<40, 256, 0, stream>>>(counts);                    // 10240
  pack_w_kernel<<<(E_ * GV) / 256, 256, 0, stream>>>(W, wp);             // 2560 blocks
  gemm_vq_kernel<<<M_ / 16, 256, 0, stream>>>(x, wp, b, u, cb, out, counts);
  entropy_kernel<<<1, 512, 0, stream>>>(counts, out);
}
